// GivensRotation_146028888592
// MI455X (gfx1250) — matrix operand order repaired and checked by reading, not compiled, not run on hardware
//
#include <hip/hip_runtime.h>

#define N_DIM  2048
#define NSTEPS 2047
#define NPAIRS 1024
#define NBATCH 8192

typedef __attribute__((ext_vector_type(16))) __bf16 v16bf;
typedef __attribute__((ext_vector_type(8)))  float  v8f;

union FragBF { unsigned int u[8]; v16bf v; };

__device__ __forceinline__ unsigned short f2bf(float f) {
  unsigned u = __float_as_uint(f);
  u += 0x7FFFu + ((u >> 16) & 1u);          // round-to-nearest-even
  return (unsigned short)(u >> 16);
}
__device__ __forceinline__ float bf2f(unsigned short h) {
  return __uint_as_float(((unsigned)h) << 16);
}
// split two f32 into packed bf16 hi parts and bf16 lo residuals
__device__ __forceinline__ void split2(float a, float b, unsigned& hi, unsigned& lo) {
  unsigned short ha = f2bf(a), hb = f2bf(b);
  unsigned short la = f2bf(a - bf2f(ha)), lb = f2bf(b - bf2f(hb));
  hi = (unsigned)ha | ((unsigned)hb << 16);
  lo = (unsigned)la | ((unsigned)lb << 16);
}

// ---------------------------------------------------------------- kernel 1
__global__ void cs_kernel(const float* __restrict__ ang,
                          float* __restrict__ c, float* __restrict__ s, int n) {
  int i = blockIdx.x * blockDim.x + threadIdx.x;
  if (i < n) {
    float sv, cv;
    sincosf(ang[i], &sv, &cv);
    c[i] = cv; s[i] = sv;
  }
}

// ---------------------------------------------------------------- kernel 2
// x (f32 row-major) -> Xhi/Xlo (bf16 row-major), one pass.
__global__ __launch_bounds__(256) void split_x_kernel(
    const float* __restrict__ X,
    unsigned short* __restrict__ Xhi, unsigned short* __restrict__ Xlo) {
  size_t i = ((size_t)blockIdx.x * 256 + threadIdx.x) * 4;
  float4 f = *(const float4*)(X + i);
  unsigned h0, l0, h1, l1;
  split2(f.x, f.y, h0, l0);
  split2(f.z, f.w, h1, l1);
  *(uint2*)(Xhi + i) = make_uint2(h0, h1);
  *(uint2*)(Xlo + i) = make_uint2(l0, l1);
}

// ---------------------------------------------------------------- kernel 3
// Each block owns 8 rows of U in LDS, laid out [col][r] (8 rows contiguous,
// 32B per column) so each Givens pair update is 4x ds_load_b128 +
// 4x ds_store_b128 instead of 32 scalar LDS ops.
__global__ __launch_bounds__(256) void build_u_kernel(
    const float* __restrict__ cbuf, const float* __restrict__ sbuf,
    const int*   __restrict__ blocks,
    unsigned short* __restrict__ Uhi, unsigned short* __restrict__ Ulo) {
  __shared__ __align__(16) float Ush[N_DIM * 8];   // 64 KB, Ush[col*8 + r]
  const int r0 = blockIdx.x * 8;

  for (int idx = threadIdx.x; idx < N_DIM * 8; idx += 256) {
    int col = idx >> 3, r = idx & 7;
    Ush[idx] = (col == r0 + r) ? 1.0f : 0.0f;
  }
  __syncthreads();

  const int2* bp = (const int2*)blocks;            // (step, pair) -> {i, j}
  for (int t = 0; t < NSTEPS; ++t) {
    const int base = t << 10;
    #pragma unroll
    for (int pp = 0; pp < 4; ++pp) {               // 1024 pairs / 256 threads
      int p = base + (pp << 8) + threadIdx.x;
      int2 ij = bp[p];
      float c = cbuf[p], s = sbuf[p];
      float* fi = &Ush[ij.x << 3];
      float* fj = &Ush[ij.y << 3];
      float4 a0 = *(float4*)fi, a1 = *(float4*)(fi + 4);
      float4 b0 = *(float4*)fj, b1 = *(float4*)(fj + 4);
      float ax[8] = {a0.x, a0.y, a0.z, a0.w, a1.x, a1.y, a1.z, a1.w};
      float bx[8] = {b0.x, b0.y, b0.z, b0.w, b1.x, b1.y, b1.z, b1.w};
      float ni[8], nj[8];
      #pragma unroll
      for (int r = 0; r < 8; ++r) {
        ni[r] = fmaf(c, ax[r],  s * bx[r]);
        nj[r] = fmaf(c, bx[r], -s * ax[r]);
      }
      *(float4*)fi       = make_float4(ni[0], ni[1], ni[2], ni[3]);
      *(float4*)(fi + 4) = make_float4(ni[4], ni[5], ni[6], ni[7]);
      *(float4*)fj       = make_float4(nj[0], nj[1], nj[2], nj[3]);
      *(float4*)(fj + 4) = make_float4(nj[4], nj[5], nj[6], nj[7]);
    }
    __syncthreads();                               // pairs change every step
  }

  // coalesced bf16 split stores: loop rows, threads sweep columns
  #pragma unroll
  for (int r = 0; r < 8; ++r) {
    for (int col = threadIdx.x; col < N_DIM; col += 256) {
      float v = Ush[(col << 3) + r];
      unsigned short h = f2bf(v);
      unsigned short l = f2bf(v - bf2f(h));
      size_t g = ((size_t)(r0 + r) << 11) + col;
      Uhi[g] = h; Ulo[g] = l;
    }
  }
}

// ------------------------------------------------------- fragment loaders
__device__ __forceinline__ v16bf load_a_frag(const unsigned short* base) {
  // A 16x32 bf16: V0..3 = 8 bf16 at base, V4..7 = 8 bf16 at base+16 elems
  uint4 q0 = *(const uint4*)base;
  uint4 q1 = *(const uint4*)(base + 16);
  FragBF f;
  f.u[0] = q0.x; f.u[1] = q0.y; f.u[2] = q0.z; f.u[3] = q0.w;
  f.u[4] = q1.x; f.u[5] = q1.y; f.u[6] = q1.z; f.u[7] = q1.w;
  return f.v;
}
__device__ __forceinline__ v16bf load_b_frag(const unsigned short* base) {
  // B 32x16 bf16: V0..3 = 8 bf16 at base, V4..7 = 8 bf16 at base+16 elems
  uint4 q0 = *(const uint4*)base;
  uint4 q1 = *(const uint4*)(base + 16);
  FragBF f;
  f.u[0] = q0.x; f.u[1] = q0.y; f.u[2] = q0.z; f.u[3] = q0.w;
  f.u[4] = q1.x; f.u[5] = q1.y; f.u[6] = q1.z; f.u[7] = q1.w;
  return f.v;
}

// ---------------------------------------------------------------- kernel 4a
// Fast path: pre-split operands; inner loop = b128 loads + WMMA only.
__global__ __launch_bounds__(256) void gemm_presplit_kernel(
    const unsigned short* __restrict__ Xhi, const unsigned short* __restrict__ Xlo,
    const unsigned short* __restrict__ Uhi, const unsigned short* __restrict__ Ulo,
    const float* __restrict__ bias, float* __restrict__ out) {
  const int lane = threadIdx.x & 31;
  const int w    = threadIdx.x >> 5;
  const int ll   = lane & 15;
  const int lh   = lane >> 4;
  const int wm   = w >> 1;
  const int wn   = w & 1;
  const int m0w  = blockIdx.x * 128 + wm * 32;
  const int n0w  = blockIdx.y * 128 + wn * 64;

  v8f acc[2][4] = {};

  for (int k0 = 0; k0 < N_DIM; k0 += 32) {
    v16bf ah[2], al[2];
    #pragma unroll
    for (int mi = 0; mi < 2; ++mi) {
      size_t off = ((size_t)(m0w + mi * 16 + ll) << 11) + (size_t)(k0 + lh * 8);
      ah[mi] = load_a_frag(Xhi + off);
      al[mi] = load_a_frag(Xlo + off);
    }
    v16bf bh[4], bl[4];
    #pragma unroll
    for (int ni = 0; ni < 4; ++ni) {
      size_t off = ((size_t)(n0w + ni * 16 + ll) << 11) + (size_t)(k0 + lh * 8);
      bh[ni] = load_b_frag(Uhi + off);
      bl[ni] = load_b_frag(Ulo + off);
    }
    #pragma unroll
    for (int mi = 0; mi < 2; ++mi)
      #pragma unroll
      for (int ni = 0; ni < 4; ++ni) {
        acc[mi][ni] = __builtin_amdgcn_wmma_f32_16x16x32_bf16(
            false, ah[mi], false, bh[ni], (short)0, acc[mi][ni], false, false);
        acc[mi][ni] = __builtin_amdgcn_wmma_f32_16x16x32_bf16(
            false, al[mi], false, bh[ni], (short)0, acc[mi][ni], false, false);
        acc[mi][ni] = __builtin_amdgcn_wmma_f32_16x16x32_bf16(
            false, ah[mi], false, bl[ni], (short)0, acc[mi][ni], false, false);
      }
  }

  #pragma unroll
  for (int mi = 0; mi < 2; ++mi)
    #pragma unroll
    for (int ni = 0; ni < 4; ++ni) {
      int n  = n0w + ni * 16 + ll;
      int mb = m0w + mi * 16 + lh * 8;
      float bv = bias[n];
      #pragma unroll
      for (int v = 0; v < 8; ++v)
        out[((size_t)(mb + v) << 11) + n] = acc[mi][ni][v] + bv;
    }
}

// ---------------------------------------------------------------- kernel 4b
// Fallback (small workspace): split x in registers inside the K-loop.
__global__ __launch_bounds__(256) void gemm_inline_kernel(
    const float* __restrict__ X,
    const unsigned short* __restrict__ Uhi, const unsigned short* __restrict__ Ulo,
    const float* __restrict__ bias, float* __restrict__ out) {
  const int lane = threadIdx.x & 31;
  const int w    = threadIdx.x >> 5;
  const int ll   = lane & 15;
  const int lh   = lane >> 4;
  const int wm   = w >> 1;
  const int wn   = w & 1;
  const int m0w  = blockIdx.x * 128 + wm * 32;
  const int n0w  = blockIdx.y * 128 + wn * 64;

  v8f acc[2][4] = {};

  for (int k0 = 0; k0 < N_DIM; k0 += 32) {
    FragBF ahi[2], alo[2];
    #pragma unroll
    for (int mi = 0; mi < 2; ++mi) {
      const float* xr = X + ((size_t)(m0w + mi * 16 + ll) << 11) + k0 + lh * 8;
      float4 f0 = ((const float4*)xr)[0];
      float4 f1 = ((const float4*)xr)[1];
      float4 f2 = ((const float4*)(xr + 16))[0];
      float4 f3 = ((const float4*)(xr + 16))[1];
      split2(f0.x, f0.y, ahi[mi].u[0], alo[mi].u[0]);
      split2(f0.z, f0.w, ahi[mi].u[1], alo[mi].u[1]);
      split2(f1.x, f1.y, ahi[mi].u[2], alo[mi].u[2]);
      split2(f1.z, f1.w, ahi[mi].u[3], alo[mi].u[3]);
      split2(f2.x, f2.y, ahi[mi].u[4], alo[mi].u[4]);
      split2(f2.z, f2.w, ahi[mi].u[5], alo[mi].u[5]);
      split2(f3.x, f3.y, ahi[mi].u[6], alo[mi].u[6]);
      split2(f3.z, f3.w, ahi[mi].u[7], alo[mi].u[7]);
    }
    v16bf bh[4], bl[4];
    #pragma unroll
    for (int ni = 0; ni < 4; ++ni) {
      size_t off = ((size_t)(n0w + ni * 16 + ll) << 11) + (size_t)(k0 + lh * 8);
      bh[ni] = load_b_frag(Uhi + off);
      bl[ni] = load_b_frag(Ulo + off);
    }
    #pragma unroll
    for (int mi = 0; mi < 2; ++mi)
      #pragma unroll
      for (int ni = 0; ni < 4; ++ni) {
        acc[mi][ni] = __builtin_amdgcn_wmma_f32_16x16x32_bf16(
            false, ahi[mi].v, false, bh[ni], (short)0, acc[mi][ni], false, false);
        acc[mi][ni] = __builtin_amdgcn_wmma_f32_16x16x32_bf16(
            false, alo[mi].v, false, bh[ni], (short)0, acc[mi][ni], false, false);
        acc[mi][ni] = __builtin_amdgcn_wmma_f32_16x16x32_bf16(
            false, ahi[mi].v, false, bl[ni], (short)0, acc[mi][ni], false, false);
      }
  }

  #pragma unroll
  for (int mi = 0; mi < 2; ++mi)
    #pragma unroll
    for (int ni = 0; ni < 4; ++ni) {
      int n  = n0w + ni * 16 + ll;
      int mb = m0w + mi * 16 + lh * 8;
      float bv = bias[n];
      #pragma unroll
      for (int v = 0; v < 8; ++v)
        out[((size_t)(mb + v) << 11) + n] = acc[mi][ni][v] + bv;
    }
}

// ---------------------------------------------------------------- launcher
extern "C" void kernel_launch(void* const* d_in, const int* in_sizes, int n_in,
                              void* d_out, int out_size, void* d_ws, size_t ws_size,
                              hipStream_t stream) {
  const float* x      = (const float*)d_in[0];   // (8192, 2048) f32
  const float* angles = (const float*)d_in[1];   // 2047*1024 f32
  const float* bias   = (const float*)d_in[2];   // (2048,) f32
  const int*   blocks = (const int*)d_in[3];     // (2047, 1024, 2) int32
  float* out = (float*)d_out;

  const size_t nang     = (size_t)NSTEPS * NPAIRS;
  const size_t cs_bytes = nang * sizeof(float);                 // 8.38 MB
  const size_t u_bytes  = (size_t)N_DIM * N_DIM * 2;            // 8.39 MB
  const size_t x_bytes  = (size_t)NBATCH * N_DIM * 2;           // 33.55 MB
  char* ws = (char*)d_ws;

  dim3 gg(NBATCH / 128, N_DIM / 128);

  if (ws_size >= 2 * x_bytes + 2 * u_bytes) {
    // fast path: [Xhi | Xlo | Uhi | Ulo]; cos/sin live inside the Xhi
    // region and die before split_x overwrites it (same-stream ordering).
    unsigned short* Xhi = (unsigned short*)ws;
    unsigned short* Xlo = (unsigned short*)(ws + x_bytes);
    unsigned short* Uhi = (unsigned short*)(ws + 2 * x_bytes);
    unsigned short* Ulo = (unsigned short*)(ws + 2 * x_bytes + u_bytes);
    float* cbuf = (float*)ws;
    float* sbuf = (float*)(ws + cs_bytes);

    cs_kernel<<<(int)(nang / 256), 256, 0, stream>>>(angles, cbuf, sbuf, (int)nang);
    build_u_kernel<<<N_DIM / 8, 256, 0, stream>>>(cbuf, sbuf, blocks, Uhi, Ulo);
    split_x_kernel<<<(int)((size_t)NBATCH * N_DIM / 4 / 256), 256, 0, stream>>>(x, Xhi, Xlo);
    gemm_presplit_kernel<<<gg, 256, 0, stream>>>(Xhi, Xlo, Uhi, Ulo, bias, out);
  } else {
    // small-workspace fallback: [cbuf | sbuf | Uhi | Ulo] (~33.5 MB)
    float* cbuf = (float*)ws;
    float* sbuf = (float*)(ws + cs_bytes);
    unsigned short* Uhi = (unsigned short*)(ws + 2 * cs_bytes);
    unsigned short* Ulo = (unsigned short*)(ws + 2 * cs_bytes + u_bytes);

    cs_kernel<<<(int)(nang / 256), 256, 0, stream>>>(angles, cbuf, sbuf, (int)nang);
    build_u_kernel<<<N_DIM / 8, 256, 0, stream>>>(cbuf, sbuf, blocks, Uhi, Ulo);
    gemm_inline_kernel<<<gg, 256, 0, stream>>>(x, Uhi, Ulo, bias, out);
  }
}
